// LSTMWithAttention_91164975825187
// MI455X (gfx1250) — matrix operand order repaired and checked by reading, not compiled, not run on hardware
//
#include <hip/hip_runtime.h>

// ---------------------------------------------------------------------------
// LSTM + attention for MI455X (gfx1250), wave32, WMMA bf16 path.
//   B=64, S=512, H=512, 4H=2048, I=84 (padded to 96 = 3 * K32)
// Layouts (all chosen so the sequential recurrence uses only wide accesses):
//   feats_bf : (S, B, 96)    bf16
//   xg_t     : (S, 4H, B)    f32   gate preactivations incl. both biases
//   h_t      : (S, H, B)     f32   LSTM hidden states for attention phase
// ---------------------------------------------------------------------------

typedef __attribute__((ext_vector_type(16))) __bf16 v16bf;
typedef __attribute__((ext_vector_type(8)))  float  v8f;

struct BfFrag { uint4 lo; uint4 hi; };   // 32 bytes == v16bf

static __device__ __forceinline__ unsigned short f32_to_bf16(float f) {
    unsigned int u = __builtin_bit_cast(unsigned int, f);
    unsigned int r = 0x7FFFu + ((u >> 16) & 1u);      // round-to-nearest-even
    return (unsigned short)((u + r) >> 16);
}

static __device__ __forceinline__ float sigmoid_f(float x) {
    return 1.0f / (1.0f + __expf(-x));
}
static __device__ __forceinline__ float tanh_f(float x) {
    float t = __expf(2.0f * x);
    return (t - 1.0f) / (t + 1.0f);
}

// 16x32 bf16 A-fragment: per lane, row m = lane&15, 16 contiguous
// k's at k0 = kt*32 + 16*hi.
static __device__ __forceinline__ v16bf load_a_frag(const unsigned short* p) {
    BfFrag f;
    f.lo = *(const uint4*)p;
    f.hi = *(const uint4*)(p + 8);
    return __builtin_bit_cast(v16bf, f);
}

#define NB 64
#define NS 512
#define NH 512
#define NG 2048
#define KPAD 96

// ---------------------------------------------------------------------------
// Phase 1a: hand-landmark normalization -> bf16 feats (S, B, 96)
// ---------------------------------------------------------------------------
__global__ __launch_bounds__(256) void prep_feats(const float* __restrict__ x,
                                                  unsigned short* __restrict__ feats_bf) {
    int idx = blockIdx.x * 256 + threadIdx.x;              // s*NB + b
    if (idx >= NB * NS) return;
    const int b = idx & (NB - 1);
    const int s = idx >> 6;
    const float* row = x + (size_t)(b * NS + s) * 266;     // 133 points * 2
    float cx = row[0], cy = row[1];
    unsigned short* op = feats_bf + (size_t)idx * KPAD;
    #pragma unroll
    for (int hnd = 0; hnd < 2; ++hnd) {
        const int p0 = hnd ? 112 : 91;
        float mnx = 3.4e38f, mny = 3.4e38f, mxx = -3.4e38f, mxy = -3.4e38f;
        for (int p = 0; p < 21; ++p) {
            float px = row[(p0 + p) * 2], py = row[(p0 + p) * 2 + 1];
            mnx = fminf(mnx, px); mxx = fmaxf(mxx, px);
            mny = fminf(mny, py); mxy = fmaxf(mxy, py);
        }
        float wx = mxx - mnx, wy = mxy - mny;
        bool ok = (wx != 0.0f) && (wy != 0.0f);
        float sx = ok ? wx : 1.0f, sy = ok ? wy : 1.0f;
        for (int p = 0; p < 21; ++p) {
            float px = row[(p0 + p) * 2], py = row[(p0 + p) * 2 + 1];
            op[hnd * 42 + p * 2 + 0] = f32_to_bf16((px - cx) / sx);
            op[hnd * 42 + p * 2 + 1] = f32_to_bf16((py - cy) / sy);
        }
    }
    for (int i = 84; i < KPAD; ++i) op[i] = 0;
}

// ---------------------------------------------------------------------------
// Phase 1b: weight conversions to bf16 (+ fused bias vector)
// ---------------------------------------------------------------------------
__global__ __launch_bounds__(256) void conv_wih(const float* __restrict__ w,
                                                const float* __restrict__ b_ih,
                                                const float* __restrict__ b_hh,
                                                unsigned short* __restrict__ o,
                                                float* __restrict__ bsum) {
    int idx = blockIdx.x * 256 + threadIdx.x;
    if (idx < NG) bsum[idx] = b_ih[idx] + b_hh[idx];
    if (idx >= NG * KPAD) return;
    int n = idx / KPAD, k = idx % KPAD;
    o[idx] = (k < 84) ? f32_to_bf16(w[n * 84 + k]) : (unsigned short)0;
}

__global__ __launch_bounds__(256) void conv_whh(const float* __restrict__ w,
                                                unsigned short* __restrict__ o) {
    int idx = blockIdx.x * 256 + threadIdx.x;
    if (idx < NG * NH) o[idx] = f32_to_bf16(w[idx]);
}

// ---------------------------------------------------------------------------
// Phase 2: xg = feats @ W_ih^T + bsum   (32768 x 96 x 2048, WMMA bf16)
// M-tile = 16 consecutive batch rows at fixed s (feats is (S,B,96)), so the
// transposed store xg_t[(s*4H + n)*B + b] is contiguous per lane (b128 x2).
// Two interleaved accumulator chains per iteration hide B-fragment latency.
// ---------------------------------------------------------------------------
__global__ __launch_bounds__(256) void xg_gemm(const unsigned short* __restrict__ feats_bf,
                                               const unsigned short* __restrict__ wih_bf,
                                               const float* __restrict__ bsum,
                                               float* __restrict__ xg_t) {
    const int mtile = blockIdx.x;                 // 0..2047; s = mtile>>2, b-base = (mtile&3)*16
    const int wave  = threadIdx.x >> 5;           // 0..7
    const int lane  = threadIdx.x & 31;
    const int hi    = lane >> 4;
    const int nl    = lane & 15;
    const int sidx  = mtile >> 2;
    const int bbase = (mtile & 3) * 16;

    // A fragments: row = mtile*16 + (lane&15); reused for all N-tiles
    v16bf a[3];
    #pragma unroll
    for (int kt = 0; kt < 3; ++kt) {
        const unsigned short* ap =
            feats_bf + (size_t)(mtile * 16 + nl) * KPAD + kt * 32 + 16 * hi;
        a[kt] = load_a_frag(ap);
    }

    for (int it = 0; it < 8; ++it) {              // constant trip count: scalar branch
        const int n0 = (it * 16 + wave) * 16 + nl;        // chain 0 column
        const int n1 = (it * 16 + 8 + wave) * 16 + nl;    // chain 1 column
        const unsigned short* bp0 = wih_bf + (size_t)n0 * KPAD + 16 * hi;
        const unsigned short* bp1 = wih_bf + (size_t)n1 * KPAD + 16 * hi;
        v8f acc0 = {}, acc1 = {};
        #pragma unroll
        for (int kt = 0; kt < 3; ++kt) {
            v16bf b0 = *(const v16bf*)(bp0 + kt * 32);
            v16bf b1 = *(const v16bf*)(bp1 + kt * 32);
            acc0 = __builtin_amdgcn_wmma_f32_16x16x32_bf16(
                false, a[kt], false, b0, (short)0, acc0, false, false);
            acc1 = __builtin_amdgcn_wmma_f32_16x16x32_bf16(
                false, a[kt], false, b1, (short)0, acc1, false, false);
        }
        float bias0 = bsum[n0];
        float bias1 = bsum[n1];
        // C layout: row m = r + 8*hi -> batch b = bbase + m : contiguous per lane
        float* op0 = xg_t + ((size_t)sidx * NG + n0) * NB + bbase + 8 * hi;
        float* op1 = xg_t + ((size_t)sidx * NG + n1) * NB + bbase + 8 * hi;
        *(float4*)(op0)     = make_float4(acc0[0] + bias0, acc0[1] + bias0,
                                          acc0[2] + bias0, acc0[3] + bias0);
        *(float4*)(op0 + 4) = make_float4(acc0[4] + bias0, acc0[5] + bias0,
                                          acc0[6] + bias0, acc0[7] + bias0);
        *(float4*)(op1)     = make_float4(acc1[0] + bias1, acc1[1] + bias1,
                                          acc1[2] + bias1, acc1[3] + bias1);
        *(float4*)(op1 + 4) = make_float4(acc1[4] + bias1, acc1[5] + bias1,
                                          acc1[6] + bias1, acc1[7] + bias1);
    }
}

// ---------------------------------------------------------------------------
// Phase 3: sequential LSTM recurrence.
// 4 blocks (16 batch rows each), 32 waves; wave w owns h columns [w*16, w*16+16).
// h double-buffered in LDS (bf16); c lives in VGPRs; W_hh streamed from L2.
// Per step per wave: 4 gate tiles x 16 K-iters = 64 v_wmma_f32_16x16x32_bf16
// over 4 independent accumulator chains; all global traffic is b128.
// ---------------------------------------------------------------------------
__global__ __launch_bounds__(1024) void lstm_recur(const unsigned short* __restrict__ whh_bf,
                                                   const float* __restrict__ xg_t,
                                                   float* __restrict__ h_t) {
    __shared__ unsigned short hbuf[2][16 * NH];   // 2 x 16KB

    const int tid  = threadIdx.x;
    const int wave = tid >> 5;                    // 0..31 == j-tile
    const int lane = tid & 31;
    const int hi   = lane >> 4;
    const int nl   = lane & 15;
    const int jcol = wave * 16 + nl;              // lane's h column, 0..511
    const int b0   = blockIdx.x * 16;             // batch-row base

    for (int i = tid; i < 16 * NH; i += 1024) hbuf[0][i] = 0;   // h0 = 0
    __syncthreads();

    // B-fragment bases: column n = W_hh row (g*512 + jcol); 16 contiguous k's
    const unsigned short* brow0 = whh_bf + ((size_t)(0 * NH + jcol)) * NH + 16 * hi;
    const unsigned short* brow1 = whh_bf + ((size_t)(1 * NH + jcol)) * NH + 16 * hi;
    const unsigned short* brow2 = whh_bf + ((size_t)(2 * NH + jcol)) * NH + 16 * hi;
    const unsigned short* brow3 = whh_bf + ((size_t)(3 * NH + jcol)) * NH + 16 * hi;

    // per-lane gate-input / h-output base offset within a (s, n)-line
    const size_t lane_off = (size_t)b0 + 8 * hi;

    float c[8];
    #pragma unroll
    for (int r = 0; r < 8; ++r) c[r] = 0.0f;

    int p = 0;
    for (int s = 0; s < NS; ++s) {
        // prefetch next step's gate lines into cache while WMMAs run
        if (s + 1 < NS) {
            #pragma unroll
            for (int g = 0; g < 4; ++g)
                __builtin_prefetch(xg_t + ((size_t)(s + 1) * NG + g * NH + jcol) * NB + lane_off, 0, 1);
        }

        v8f acc0 = {}, acc1 = {}, acc2 = {}, acc3 = {};
        const unsigned short* arow = &hbuf[p][nl * NH] + 16 * hi;   // A row = lane&15
        #pragma unroll
        for (int kt = 0; kt < NH / 32; ++kt) {
            // group all fragment loads so the clause former batches them
            v16bf a   = load_a_frag(arow + kt * 32);
            v16bf bb0 = *(const v16bf*)(brow0 + kt * 32);
            v16bf bb1 = *(const v16bf*)(brow1 + kt * 32);
            v16bf bb2 = *(const v16bf*)(brow2 + kt * 32);
            v16bf bb3 = *(const v16bf*)(brow3 + kt * 32);
            acc0 = __builtin_amdgcn_wmma_f32_16x16x32_bf16(false, a, false, bb0, (short)0, acc0, false, false);
            acc1 = __builtin_amdgcn_wmma_f32_16x16x32_bf16(false, a, false, bb1, (short)0, acc1, false, false);
            acc2 = __builtin_amdgcn_wmma_f32_16x16x32_bf16(false, a, false, bb2, (short)0, acc2, false, false);
            acc3 = __builtin_amdgcn_wmma_f32_16x16x32_bf16(false, a, false, bb3, (short)0, acc3, false, false);
        }

        // gate inputs: contiguous 8 floats per lane per gate (2 x b128 each)
        float xv[4][8];
        #pragma unroll
        for (int g = 0; g < 4; ++g) {
            const float* gp = xg_t + ((size_t)s * NG + g * NH + jcol) * NB + lane_off;
            float4 lo4 = *(const float4*)(gp);
            float4 hi4 = *(const float4*)(gp + 4);
            xv[g][0] = lo4.x; xv[g][1] = lo4.y; xv[g][2] = lo4.z; xv[g][3] = lo4.w;
            xv[g][4] = hi4.x; xv[g][5] = hi4.y; xv[g][6] = hi4.z; xv[g][7] = hi4.w;
        }

        // nonlinearity + state update; C layout: M = r + 8*hi, N = lane&15
        float hv[8];
        #pragma unroll
        for (int r = 0; r < 8; ++r) {
            const int m = r + 8 * hi;
            float gi = sigmoid_f(acc0[r] + xv[0][r]);
            float gf = sigmoid_f(acc1[r] + xv[1][r]);
            float gg = tanh_f  (acc2[r] + xv[2][r]);
            float go = sigmoid_f(acc3[r] + xv[3][r]);
            c[r] = gf * c[r] + gi * gg;
            hv[r] = go * tanh_f(c[r]);
            hbuf[p ^ 1][m * NH + jcol] = f32_to_bf16(hv[r]);
        }
        float* hp = h_t + ((size_t)s * NH + jcol) * NB + lane_off;
        *(float4*)(hp)     = make_float4(hv[0], hv[1], hv[2], hv[3]);
        *(float4*)(hp + 4) = make_float4(hv[4], hv[5], hv[6], hv[7]);

        __syncthreads();
        p ^= 1;
    }
}

// ---------------------------------------------------------------------------
// Phase 4: attention scores -> softmax(S) -> context -> FC.  One block per b.
// h_t is (S, H, B).  d_out: [0,6400) logits (64x100), [6400,39168) weights.
// ---------------------------------------------------------------------------
__global__ __launch_bounds__(256) void attn_fc(const float* __restrict__ h_t,
                                               const float* __restrict__ attn_w,
                                               const float* __restrict__ fc_w,
                                               const float* __restrict__ fc_b,
                                               float* __restrict__ out) {
    __shared__ float sc[NS];
    __shared__ float red[NS];
    __shared__ float ctx[NH];
    const int b = blockIdx.x, tid = threadIdx.x;

    for (int s = tid; s < NS; s += 256) {
        float d = 0.0f;
        const float* hr = h_t + (size_t)s * NH * NB + b;
        for (int k = 0; k < NH; ++k) d += hr[(size_t)k * NB] * attn_w[k];
        sc[s] = d;
    }
    __syncthreads();
    for (int s = tid; s < NS; s += 256) red[s] = sc[s];
    __syncthreads();
    for (int off = 256; off > 0; off >>= 1) {
        if (tid < off && tid + off < NS) red[tid] = fmaxf(red[tid], red[tid + off]);
        __syncthreads();
    }
    float mx = red[0];
    __syncthreads();
    for (int s = tid; s < NS; s += 256) sc[s] = __expf(sc[s] - mx);
    __syncthreads();
    for (int s = tid; s < NS; s += 256) red[s] = sc[s];
    __syncthreads();
    for (int off = 256; off > 0; off >>= 1) {
        if (tid < off && tid + off < NS) red[tid] += red[tid + off];
        __syncthreads();
    }
    float inv = 1.0f / red[0];
    __syncthreads();
    for (int s = tid; s < NS; s += 256) {
        float w = sc[s] * inv;
        sc[s] = w;
        out[NB * 100 + b * NS + s] = w;             // weights output
    }
    __syncthreads();
    for (int hc = tid; hc < NH; hc += 256) {
        float acc = 0.0f;
        const float* hr = h_t + (size_t)hc * NB + b;
        for (int s = 0; s < NS; ++s) acc += sc[s] * hr[(size_t)s * NH * NB];
        ctx[hc] = acc;
    }
    __syncthreads();
    if (tid < 100) {
        float acc = fc_b[tid];
        const float* fr = fc_w + (size_t)tid * NH;
        for (int k = 0; k < NH; ++k) acc += ctx[k] * fr[k];
        out[b * 100 + tid] = acc;                   // logits output
    }
}

// ---------------------------------------------------------------------------
extern "C" void kernel_launch(void* const* d_in, const int* in_sizes, int n_in,
                              void* d_out, int out_size, void* d_ws, size_t ws_size,
                              hipStream_t stream) {
    const float* x      = (const float*)d_in[0];
    const float* W_ih   = (const float*)d_in[1];
    const float* W_hh   = (const float*)d_in[2];
    const float* b_ih   = (const float*)d_in[3];
    const float* b_hh   = (const float*)d_in[4];
    const float* attn_w = (const float*)d_in[5];
    const float* fc_w   = (const float*)d_in[6];
    const float* fc_b   = (const float*)d_in[7];
    float* out = (float*)d_out;
    (void)in_sizes; (void)n_in; (void)out_size; (void)ws_size;

    char* ws = (char*)d_ws;
    size_t off = 0;
    auto take = [&](size_t bytes) -> char* {
        char* p = ws + off;
        off = (off + bytes + 255) & ~(size_t)255;
        return p;
    };
    unsigned short* feats_bf = (unsigned short*)take((size_t)NB * NS * KPAD * 2); //  6.3 MB
    unsigned short* wih_bf   = (unsigned short*)take((size_t)NG * KPAD * 2);      //  0.4 MB
    unsigned short* whh_bf   = (unsigned short*)take((size_t)NG * NH * 2);        //  2.1 MB
    float*          bsum     = (float*)take((size_t)NG * 4);                      //  8 KB
    float*          xg_t     = (float*)take((size_t)NS * NG * NB * 4);            // 268 MB
    float*          h_t      = (float*)take((size_t)NS * NH * NB * 4);            //  67 MB

    prep_feats<<<(NB * NS + 255) / 256, 256, 0, stream>>>(x, feats_bf);
    conv_wih<<<(NG * KPAD + 255) / 256, 256, 0, stream>>>(W_ih, b_ih, b_hh, wih_bf, bsum);
    conv_whh<<<(NG * NH + 255) / 256, 256, 0, stream>>>(W_hh, whh_bf);
    xg_gemm<<<(NB * NS) / 16, 256, 0, stream>>>(feats_bf, wih_bf, bsum, xg_t);
    lstm_recur<<<NB / 16, 1024, 0, stream>>>(whh_bf, xg_t, h_t);
    attn_fc<<<NB, 256, 0, stream>>>(h_t, attn_w, fc_w, fc_b, out);
}
